// LSTM_Extraction_Layer_36146444763629
// MI455X (gfx1250) — compile-verified, matrix operand order checked by reading, not run on hardware
//
#include <hip/hip_runtime.h>
#include <hip/hip_bf16.h>

// ---------------------------------------------------------------------------
// Types for CDNA5 WMMA (wave32): A/B = 16 x bf16 per lane, C/D = 8 x f32.
// ---------------------------------------------------------------------------
typedef __attribute__((ext_vector_type(16))) __bf16 v16bf;
typedef __attribute__((ext_vector_type(8)))  float  v8f;

#define WMMA_BF16(a, b, c) \
    __builtin_amdgcn_wmma_f32_16x16x32_bf16(false, (a), false, (b), (short)0, (c), false, false)

// ---------------------------------------------------------------------------
// fp32 -> bf16 convert (weights)
// ---------------------------------------------------------------------------
__global__ void f32_to_bf16_k(const float* __restrict__ in, __bf16* __restrict__ out, size_t n) {
    size_t i = (size_t)blockIdx.x * blockDim.x + threadIdx.x;
    if (i < n) out[i] = (__bf16)in[i];
}

// (B,T,E) fp32 -> (T,B,E) bf16 (transpose + convert for time-major GEMM rows)
__global__ void xpose_bf16_k(const float* __restrict__ in, __bf16* __restrict__ out,
                             int B, int T, int E) {
    size_t i = (size_t)blockIdx.x * blockDim.x + threadIdx.x;
    size_t n = (size_t)B * T * E;
    if (i >= n) return;
    int e  = (int)(i % E);
    size_t bt = i / E;
    int t = (int)(bt % T);
    int b = (int)(bt / T);
    out[((size_t)t * B + b) * E + e] = (__bf16)in[i];
}

// ---------------------------------------------------------------------------
// WMMA GEMM: OUT[M,N] = A[M,K](bf16) @ W[N,K]^T(bf16) + bias[N]   (fp32 out)
// One wave computes a 16x64 strip: one A fragment feeds 4 WMMAs per K step
// (4x reuse of the dominant activation stream). K fully unrolled.
// ---------------------------------------------------------------------------
template<int K>
__global__ __launch_bounds__(256)
void gemm_bias_k(const __bf16* __restrict__ A, const __bf16* __restrict__ W,
                 const float* __restrict__ bias, float* __restrict__ OUT,
                 int M, int N) {
    const int lane   = threadIdx.x & 31;
    const int wave_g = blockIdx.x * (blockDim.x >> 5) + (threadIdx.x >> 5);
    const int nsn    = N >> 6;               // 64-wide N strips
    const int tm     = wave_g / nsn;
    const int ns     = wave_g % nsn;
    if (tm * 16 >= M) return;

    const int lm    = lane & 15;
    const int khalf = lane >> 4;

    const __bf16* arow = A + (size_t)(tm * 16 + lm) * K;
    const __bf16* w0   = W + (size_t)(ns * 64 + lm) * K;   // rows ns*64+lm + {0,16,32,48}

    v8f acc0 = {}, acc1 = {}, acc2 = {}, acc3 = {};
#pragma unroll
    for (int kk = 0; kk < K / 32; ++kk) {
        const int ko = kk * 32 + khalf * 16;
        v16bf a  = *(const v16bf*)(arow + ko);
        v16bf b0 = *(const v16bf*)(w0 + (size_t) 0 * K + ko);
        v16bf b1 = *(const v16bf*)(w0 + (size_t)16 * K + ko);
        v16bf b2 = *(const v16bf*)(w0 + (size_t)32 * K + ko);
        v16bf b3 = *(const v16bf*)(w0 + (size_t)48 * K + ko);
        acc0 = WMMA_BF16(a, b0, acc0);
        acc1 = WMMA_BF16(a, b1, acc1);
        acc2 = WMMA_BF16(a, b2, acc2);
        acc3 = WMMA_BF16(a, b3, acc3);
    }
#pragma unroll
    for (int q = 0; q < 4; ++q) {
        const v8f acc = (q == 0) ? acc0 : (q == 1) ? acc1 : (q == 2) ? acc2 : acc3;
        const int ncol = ns * 64 + q * 16 + lm;
        const float bn = bias[ncol];
#pragma unroll
        for (int r = 0; r < 8; ++r) {
            // C/D layout: VGPR r -> row (r + 8*khalf), col = lane&15
            OUT[(size_t)(tm * 16 + r + 8 * khalf) * N + ncol] = acc[r] + bn;
        }
    }
}

// ---------------------------------------------------------------------------
// Persistent LSTM recurrence. One workgroup (512 threads = 16 waves) owns a
// 16-row batch tile for the full time loop. h (bf16) and c (fp32) live in LDS.
// STAGE_WHH=true (H=128): Whh (128KB bf16) is staged into LDS once via
// global_load_async_to_lds_b128 (ASYNCcnt) before the time loop; all per-step
// B fragments then come from LDS. H=256 (512KB) streams Whh from L2 instead.
//   xg : (T, B, 4H) fp32 preactivations (already has Wih@x + b)
//   whh: (4H, H) bf16
//   out: (T, B, out_stride) fp32; h written at column offset col_off
// ---------------------------------------------------------------------------
template<int H, bool STAGE_WHH>
__global__ __launch_bounds__(512)
void lstm_rec_k(const float* __restrict__ xg, const __bf16* __restrict__ whh,
                float* __restrict__ out, int T, int B,
                int out_stride, int col_off, int reverse) {
    constexpr int N4 = 4 * H;
    constexpr int NT = N4 / 16;   // N tiles of the gate matrix
    constexpr int KI = H / 32;    // WMMA K iterations
    constexpr int NW = 16;        // waves per workgroup

    extern __shared__ char smem[];
    float*  g_lds   = (float*)smem;                 // [16][4H] gate preactivations
    float*  c_lds   = g_lds + 16 * N4;              // [16][H]  cell state
    __bf16* h_lds   = (__bf16*)(c_lds + 16 * H);    // [16][H]  hidden (bf16 WMMA A)
    __bf16* whh_lds = h_lds + 16 * H;               // [4H][H]  staged weights (opt)

    const int b0   = blockIdx.x * 16;
    const int tid  = threadIdx.x;
    const int wave = tid >> 5;
    const int lane = tid & 31;
    const int lm    = lane & 15;
    const int khalf = lane >> 4;

    for (int i = tid; i < 16 * H; i += 512) {
        h_lds[i] = (__bf16)0.0f;
        c_lds[i] = 0.0f;
    }

    if constexpr (STAGE_WHH) {
        // Async DMA Whh -> LDS, 16B per lane per issue; tracked by ASYNCcnt.
        const unsigned lds_base = (unsigned)(size_t)(void*)whh_lds;
        const char*    gbase    = (const char*)whh;
        constexpr int  BYTES    = N4 * H * 2;
        for (int i = tid * 16; i < BYTES; i += 512 * 16) {
            unsigned           laddr = lds_base + (unsigned)i;
            unsigned long long gaddr = (unsigned long long)(gbase + i);
            asm volatile("global_load_async_to_lds_b128 %0, %1, off"
                         :: "v"(laddr), "v"(gaddr) : "memory");
        }
        asm volatile("s_wait_asynccnt 0x0" ::: "memory");
    }
    __syncthreads();

    for (int step = 0; step < T; ++step) {
        const int t = reverse ? (T - 1 - step) : step;
        const float* xgrow = xg + ((size_t)t * B + b0) * N4;

        // --- gate GEMM: g = xg[t] + h @ whh^T ---
        for (int tn = wave; tn < NT; tn += NW) {
            const int n = tn * 16 + lm;
            v8f acc;
#pragma unroll
            for (int r = 0; r < 8; ++r)
                acc[r] = xgrow[(size_t)(r + 8 * khalf) * N4 + tn * 16 + lm];
            if constexpr (STAGE_WHH) {
                const __bf16* wr = whh_lds + (size_t)n * H;
#pragma unroll
                for (int kk = 0; kk < KI; ++kk) {
                    v16bf a = *(const v16bf*)(h_lds + lm * H + kk * 32 + khalf * 16);
                    v16bf b = *(const v16bf*)(wr + kk * 32 + khalf * 16);
                    acc = WMMA_BF16(a, b, acc);
                }
            } else {
                const __bf16* wr = whh + (size_t)n * H;
                __builtin_prefetch(wr, 0, 1);   // global_prefetch: keep L2-resident Whh hot
#pragma unroll
                for (int kk = 0; kk < KI; ++kk) {
                    v16bf a = *(const v16bf*)(h_lds + lm * H + kk * 32 + khalf * 16);
                    v16bf b = *(const v16bf*)(wr + kk * 32 + khalf * 16);
                    acc = WMMA_BF16(a, b, acc);
                }
            }
#pragma unroll
            for (int r = 0; r < 8; ++r)
                g_lds[(r + 8 * khalf) * N4 + tn * 16 + lm] = acc[r];
        }
        __syncthreads();

        // --- elementwise gates: c,h update ---
        for (int i = tid; i < 16 * H; i += 512) {
            const int row = i / H;
            const int hc  = i % H;
            const float gi = g_lds[row * N4 + hc];
            const float gf = g_lds[row * N4 + hc + H];
            const float gg = g_lds[row * N4 + hc + 2 * H];
            const float go = g_lds[row * N4 + hc + 3 * H];
            const float si = 1.0f / (1.0f + __expf(-gi));
            const float sf = 1.0f / (1.0f + __expf(-gf));
            const float so = 1.0f / (1.0f + __expf(-go));
            float c = sf * c_lds[row * H + hc] + si * tanhf(gg);
            float h = so * tanhf(c);
            c_lds[row * H + hc] = c;
            h_lds[row * H + hc] = (__bf16)h;
            out[((size_t)t * B + b0 + row) * out_stride + col_off + hc] = h;
        }
        __syncthreads();
    }
}

// ---------------------------------------------------------------------------
// LayerNorm over 256 features, one wave32 per row (8 values/lane, shuffle
// butterfly reduce). Optional residual input; writes fp32 + bf16 copies.
// ---------------------------------------------------------------------------
__global__ __launch_bounds__(256)
void layernorm_k(const float* __restrict__ X, const float* __restrict__ RES,
                 const float* __restrict__ g, const float* __restrict__ b,
                 float* __restrict__ OUT_F, __bf16* __restrict__ OUT_BF, int rows) {
    const int row = blockIdx.x * 8 + (threadIdx.x >> 5);
    if (row >= rows) return;
    const int lane = threadIdx.x & 31;
    const float* xr = X + (size_t)row * 256;
    const float* rr = RES ? RES + (size_t)row * 256 : nullptr;

    float v[8];
    float s = 0.0f;
#pragma unroll
    for (int j = 0; j < 8; ++j) {
        float x = xr[lane + j * 32];
        if (rr) x += rr[lane + j * 32];
        v[j] = x;
        s += x;
    }
#pragma unroll
    for (int off = 16; off > 0; off >>= 1) s += __shfl_xor(s, off, 32);
    const float mu = s * (1.0f / 256.0f);
    float s2 = 0.0f;
#pragma unroll
    for (int j = 0; j < 8; ++j) { float d = v[j] - mu; s2 += d * d; }
#pragma unroll
    for (int off = 16; off > 0; off >>= 1) s2 += __shfl_xor(s2, off, 32);
    const float rstd = __frsqrt_rn(s2 * (1.0f / 256.0f) + 1e-5f);
#pragma unroll
    for (int j = 0; j < 8; ++j) {
        const int col = lane + j * 32;
        const float y = (v[j] - mu) * rstd * g[col] + b[col];
        OUT_F[(size_t)row * 256 + col]  = y;
        OUT_BF[(size_t)row * 256 + col] = (__bf16)y;
    }
}

// ---------------------------------------------------------------------------
// Final gather at last_idx + affine (BN-inference) + ReLU
// ---------------------------------------------------------------------------
__global__ __launch_bounds__(256)
void head_k(const float* __restrict__ L2, const long long* __restrict__ idx,
            const float* __restrict__ g, const float* __restrict__ bb,
            float* __restrict__ out, int B) {
    const int b = blockIdx.x;
    const int j = threadIdx.x;
    const long long t = idx[b];
    const float scale = rsqrtf(1.0f + 1e-5f);
    const float y = L2[((size_t)t * B + b) * 256 + j] * scale * g[j] + bb[j];
    out[(size_t)b * 256 + j] = fmaxf(y, 0.0f);
}

// ---------------------------------------------------------------------------
// Host driver
// ---------------------------------------------------------------------------
extern "C" void kernel_launch(void* const* d_in, const int* in_sizes, int n_in,
                              void* d_out, int out_size, void* d_ws, size_t ws_size,
                              hipStream_t stream) {
    (void)in_sizes; (void)n_in; (void)out_size; (void)ws_size;
    constexpr int B = 256, T = 256, E = 128, H1 = 128, H2 = 256;
    constexpr size_t MR = (size_t)B * T;   // 65536 rows

    const float*     inp_embed = (const float*)d_in[0];
    const long long* last_idx  = (const long long*)d_in[1];
    const float* Wih_f = (const float*)d_in[2];
    const float* Whh_f = (const float*)d_in[3];
    const float* b_f   = (const float*)d_in[4];
    const float* Wih_r = (const float*)d_in[5];
    const float* Whh_r = (const float*)d_in[6];
    const float* b_r   = (const float*)d_in[7];
    const float* ln1_g = (const float*)d_in[8];
    const float* ln1_b = (const float*)d_in[9];
    const float* W1ih  = (const float*)d_in[10];
    const float* W1hh  = (const float*)d_in[11];
    const float* b1    = (const float*)d_in[12];
    const float* ln2_g = (const float*)d_in[13];
    const float* ln2_b = (const float*)d_in[14];
    const float* W2ih  = (const float*)d_in[15];
    const float* W2hh  = (const float*)d_in[16];
    const float* b2    = (const float*)d_in[17];
    const float* bn_g  = (const float*)d_in[18];
    const float* bn_b  = (const float*)d_in[19];
    float* out = (float*)d_out;

    // ---- workspace bump layout (region o_xg reused for all 3 layers' gates) ----
    char* ws = (char*)d_ws;
    size_t off = 0;
    auto alloc = [&](size_t bytes) { size_t o = off; off += (bytes + 255) & ~(size_t)255; return o; };
    const size_t o_xg   = alloc(MR * 1024 * sizeof(float));   // 256MB: xg_f|xg_r, then xg1, xg2
    const size_t o_bi   = alloc(MR * 256 * sizeof(float));    // bilstm concat / LN1 (in place)
    const size_t o_bibf = alloc(MR * 256 * sizeof(__bf16));
    const size_t o_l1   = alloc(MR * 256 * sizeof(float));    // lstm1 out / LN2 (in place)
    const size_t o_l1bf = alloc(MR * 256 * sizeof(__bf16));
    const size_t o_l2   = alloc(MR * 256 * sizeof(float));    // lstm2 out
    const size_t o_xbf  = alloc(MR * E * sizeof(__bf16));     // time-major bf16 embeddings
    const size_t o_w    = alloc((4 * 65536 + 4 * 262144) * sizeof(__bf16));

    float*  xg_f = (float*)(ws + o_xg);
    float*  xg_r = xg_f + MR * 512;
    float*  xg12 = xg_f;                 // reused for layers 2/3
    float*  bi   = (float*)(ws + o_bi);
    __bf16* bibf = (__bf16*)(ws + o_bibf);
    float*  l1   = (float*)(ws + o_l1);
    __bf16* l1bf = (__bf16*)(ws + o_l1bf);
    float*  l2   = (float*)(ws + o_l2);
    __bf16* xbf  = (__bf16*)(ws + o_xbf);
    __bf16* wbf  = (__bf16*)(ws + o_w);
    __bf16* wihf = wbf;
    __bf16* whhf = wihf + 65536;
    __bf16* wihr = whhf + 65536;
    __bf16* whhr = wihr + 65536;
    __bf16* w1ih = whhr + 65536;
    __bf16* w1hh = w1ih + 262144;
    __bf16* w2ih = w1hh + 262144;
    __bf16* w2hh = w2ih + 262144;

    // ---- 0) precision conversion ----
    {
        size_t n = MR * E;
        xpose_bf16_k<<<(unsigned)((n + 255) / 256), 256, 0, stream>>>(inp_embed, xbf, B, T, E);
        auto cvt = [&](const float* src, __bf16* dst, size_t cnt) {
            f32_to_bf16_k<<<(unsigned)((cnt + 255) / 256), 256, 0, stream>>>(src, dst, cnt);
        };
        cvt(Wih_f, wihf, 65536);  cvt(Whh_f, whhf, 65536);
        cvt(Wih_r, wihr, 65536);  cvt(Whh_r, whhr, 65536);
        cvt(W1ih, w1ih, 262144);  cvt(W1hh, w1hh, 262144);
        cvt(W2ih, w2ih, 262144);  cvt(W2hh, w2hh, 262144);
    }

    auto gemm_blocks = [&](int M, int N) { return (unsigned)(((size_t)(M / 16) * (N / 64)) / 8); };
    // LDS: g=16*4H*4, c=16*H*4, h=16*H*2 (+ staged Whh=4H*H*2 for H=128)
    const size_t smem1 = 16 * 4 * H1 * 4 + 16 * H1 * 4 + 16 * H1 * 2 + (size_t)4 * H1 * H1 * 2;
    const size_t smem2 = 16 * 4 * H2 * 4 + 16 * H2 * 4 + 16 * H2 * 2;

    // ---- 1) BiLSTM layer (H=128, Whh staged into LDS via async DMA) ----
    gemm_bias_k<128><<<gemm_blocks((int)MR, 512), 256, 0, stream>>>(xbf, wihf, b_f, xg_f, (int)MR, 512);
    gemm_bias_k<128><<<gemm_blocks((int)MR, 512), 256, 0, stream>>>(xbf, wihr, b_r, xg_r, (int)MR, 512);
    lstm_rec_k<128, true><<<B / 16, 512, (unsigned)smem1, stream>>>(xg_f, whhf, bi, T, B, 256, 0,   0);
    lstm_rec_k<128, true><<<B / 16, 512, (unsigned)smem1, stream>>>(xg_r, whhr, bi, T, B, 256, 128, 1);
    layernorm_k<<<(unsigned)(MR / 8), 256, 0, stream>>>(bi, nullptr, ln1_g, ln1_b, bi, bibf, (int)MR);

    // ---- 2) LSTM layer 2 (H=256, Whh streams from 192MB L2) ----
    gemm_bias_k<256><<<gemm_blocks((int)MR, 1024), 256, 0, stream>>>(bibf, w1ih, b1, xg12, (int)MR, 1024);
    lstm_rec_k<256, false><<<B / 16, 512, (unsigned)smem2, stream>>>(xg12, w1hh, l1, T, B, 256, 0, 0);
    layernorm_k<<<(unsigned)(MR / 8), 256, 0, stream>>>(l1, bi, ln2_g, ln2_b, l1, l1bf, (int)MR);

    // ---- 3) LSTM layer 3 ----
    gemm_bias_k<256><<<gemm_blocks((int)MR, 1024), 256, 0, stream>>>(l1bf, w2ih, b2, xg12, (int)MR, 1024);
    lstm_rec_k<256, false><<<B / 16, 512, (unsigned)smem2, stream>>>(xg12, w2hh, l2, T, B, 256, 0, 0);

    // ---- 4) gather + affine + relu ----
    head_k<<<B, 256, 0, stream>>>(l2, last_idx, bn_g, bn_b, out, B);
}
